// VQVAE_79302276153480
// MI455X (gfx1250) — matrix operand order repaired and checked by reading, not compiled, not run on hardware
//
#include <hip/hip_runtime.h>
#include <stdint.h>

// ---------------------------------------------------------------------------
// VQ-VAE forward for MI455X (gfx1250, wave32, WMMA 16x16x32 f16 -> f32 acc)
// - NHWC f16 activations: im2col rows contiguous -> B tiles staged with
//   global_load_async_to_lds_b128 (ASYNCcnt), double-buffered so DMA of
//   chunk i+1 overlaps WMMA of chunk i (s_wait_asynccnt <= in-flight).
// - Weights packed in WMMA fragment order: A fragment = one 32B contiguous
//   load per lane (2x global_load_b128).
// - Block tile M64 (4 waves) x N64 (4 subtiles): 4 v_wmma per A fragment.
// - Transposed convs parity-decomposed (K = Ci*4, 4x fewer FLOPs).
// ---------------------------------------------------------------------------

typedef __attribute__((ext_vector_type(16))) _Float16 v16h;
typedef __attribute__((ext_vector_type(8)))  _Float16 v8h;
typedef __attribute__((ext_vector_type(8)))  float    v8f;

#define BLK 128   // 4 waves of 32
#define KT  32    // K chunk (WMMA K)

// A-fragment K mapping (16-bit A 16x32, contiguous: k = 16*(lane/16) + j)
__device__ __forceinline__ int kmapA(int lane, int j) {
  return j + ((lane >= 16) ? 16 : 0);
}

__device__ __forceinline__ void async_b128(uint32_t lds_off, const void* g) {
  asm volatile("global_load_async_to_lds_b128 %0, %1, off"
               :: "v"(lds_off), "v"(g) : "memory");
}
template<int N>
__device__ __forceinline__ void wait_async() {
  if constexpr (N == 0)      asm volatile("s_wait_asynccnt 0x0" ::: "memory");
  else if constexpr (N == 1) asm volatile("s_wait_asynccnt 0x1" ::: "memory");
  else if constexpr (N == 2) asm volatile("s_wait_asynccnt 0x2" ::: "memory");
  else                       asm volatile("s_wait_asynccnt 0x8" ::: "memory");
}

// ---------------------------------------------------------------------------
// Direct conv, NHWC, implicit GEMM over flattened (n,oh,ow).
// K order = (kh,kw,ci), CiPad = 1<<ciSh.  Kpad == KH*KW*CiPad.
// ASYNC_B: double-buffered async-LDS staging (CiPad==CiReal, no input relu).
// ---------------------------------------------------------------------------
template<int KH, int KW, int STRIDE, int PAD, bool ASYNC_B, bool RELU_IN,
         bool BIAS, bool RELU_OUT, bool ADD_RES, bool STORE_F16, bool OUT_F32>
__global__ __launch_bounds__(BLK) void conv_wmma(
    const _Float16* __restrict__ inh, const _Float16* __restrict__ Afrag,
    const float* __restrict__ bias, const _Float16* __restrict__ resid,
    _Float16* __restrict__ outh, float* __restrict__ outf,
    int CiReal, int ciSh, int Co, int Hi, int Wi, int Ho, int Wo, int Kpad,
    int hwSh, int woSh)
{
  __shared__ _Float16 ldsB[2 * 64 * KT];       // ping-pong, 2 x 4 KB
  const int tid  = threadIdx.x;
  const int lane = tid & 31;
  const int wv   = tid >> 5;

  const int p0 = blockIdx.x * 64;              // first flattened output pos

  const int mt      = blockIdx.y * 4 + wv;
  const int mTiles  = (Co + 15) / 16;
  const bool mvalid = (mt < mTiles);
  const int co_base = mt * 16;
  const int nCh     = Kpad >> 5;

  const int nn = lane & 15;
  const _Float16* afrag = Afrag + (size_t)mt * nCh * 512 + lane * 16;

  // staging coords: thread owns position nnS
  const int nnS  = tid >> 1;
  const int posS = p0 + nnS;
  const int nS   = posS >> hwSh;
  const int remS = posS & ((1 << hwSh) - 1);
  const int ohS  = remS >> woSh;
  const int owS  = remS & ((1 << woSh) - 1);

  v8f c[4] = {{}, {}, {}, {}};

  if constexpr (ASYNC_B) {
    const int ciMask  = (1 << ciSh) - 1;
    const int segBase = (tid & 1) * 2;
    auto issueChunk = [&](int ck, int bsel) -> bool {
      const int kb  = ck * KT;
      const int tap = kb >> ciSh;
      const int kh  = tap / KW;
      const int kw2 = tap - kh * KW;
      const int ci0 = kb & ciMask;
      const int ih  = ohS * STRIDE - PAD + kh;
      const int iw  = owS * STRIDE - PAD + kw2;
      const bool ok = (ih >= 0 && ih < Hi && iw >= 0 && iw < Wi);
      const int ihc = ok ? ih : 0;               // clamp: always issue 2
      const int iwc = ok ? iw : 0;
      const _Float16* src =
          inh + (((size_t)nS * Hi + ihc) * Wi + iwc) * CiReal + ci0;
      _Float16* dstRow = &ldsB[bsel * (64 * KT) + nnS * KT];
      #pragma unroll
      for (int u = 0; u < 2; ++u) {
        const int seg = segBase + u;
        async_b128((uint32_t)(uintptr_t)(dstRow + seg * 8),
                   (const void*)(src + seg * 8));
      }
      return ok;
    };

    bool okCur = issueChunk(0, 0);
    for (int cIdx = 0; cIdx < nCh; ++cIdx) {
      const int cur = cIdx & 1;
      bool okNext = true;
      if (cIdx + 1 < nCh) { okNext = issueChunk(cIdx + 1, cur ^ 1); wait_async<2>(); }
      else                { wait_async<0>(); }
      if (!okCur) {
        v8h zz = {};
        #pragma unroll
        for (int u = 0; u < 2; ++u)
          *(v8h*)(&ldsB[cur * (64 * KT) + nnS * KT + (segBase + u) * 8]) = zz;
      }
      __syncthreads();
      if (mvalid) {
        const v16h a = *(const v16h*)(afrag + (size_t)cIdx * 512);
        const int bo = (lane >= 16) ? 16 : 0;
        #pragma unroll
        for (int s = 0; s < 4; ++s) {
          v16h b;
          const _Float16* brow = &ldsB[cur * (64 * KT) + (s * 16 + nn) * KT + bo];
          #pragma unroll
          for (int j = 0; j < 16; ++j) b[j] = brow[j];
          c[s] = __builtin_amdgcn_wmma_f32_16x16x32_f16(false, a, false, b,
                                                        (short)0, c[s], false, false);
        }
      }
      __syncthreads();
      okCur = okNext;
    }
  } else {
    // VALU gather path (input relu and/or padded Ci), single buffer
    const int kk0 = (tid & 1) * 16;
    for (int cIdx = 0; cIdx < nCh; ++cIdx) {
      const int kb = cIdx * KT;
      __syncthreads();
      #pragma unroll
      for (int q = 0; q < 16; ++q) {
        const int K   = kb + kk0 + q;
        const int tap = K >> ciSh;
        const int ci  = K & ((1 << ciSh) - 1);
        const int kh  = tap / KW;
        const int kw2 = tap - kh * KW;
        const int ih  = ohS * STRIDE - PAD + kh;
        const int iw  = owS * STRIDE - PAD + kw2;
        _Float16 val = (_Float16)0.f;
        if (ci < CiReal && ih >= 0 && ih < Hi && iw >= 0 && iw < Wi) {
          float xv = (float)inh[(((size_t)nS * Hi + ih) * Wi + iw) * CiReal + ci];
          if (RELU_IN) xv = fmaxf(xv, 0.f);
          val = (_Float16)xv;
        }
        ldsB[nnS * KT + kk0 + q] = val;
      }
      __syncthreads();
      if (mvalid) {
        const v16h a = *(const v16h*)(afrag + (size_t)cIdx * 512);
        const int bo = (lane >= 16) ? 16 : 0;
        #pragma unroll
        for (int s = 0; s < 4; ++s) {
          v16h b;
          const _Float16* brow = &ldsB[(s * 16 + nn) * KT + bo];
          #pragma unroll
          for (int j = 0; j < 16; ++j) b[j] = brow[j];
          c[s] = __builtin_amdgcn_wmma_f32_16x16x32_f16(false, a, false, b,
                                                        (short)0, c[s], false, false);
        }
      }
    }
  }

  if (!mvalid) return;
  const int rb = (lane >= 16) ? 8 : 0;
  #pragma unroll
  for (int s = 0; s < 4; ++s) {
    const int pos = p0 + s * 16 + nn;
    #pragma unroll
    for (int r = 0; r < 8; ++r) {
      const int com = co_base + rb + r;
      if (com < Co) {
        float v = c[s][r];
        if (BIAS) v += bias[com];
        const size_t oidx = (size_t)pos * Co + com;   // NHWC
        if (ADD_RES) v += (float)resid[oidx];
        if (RELU_OUT) v = fmaxf(v, 0.f);
        if (STORE_F16) outh[oidx] = (_Float16)v;
        if (OUT_F32)  outf[oidx] = v;                 // NHWC f32 (z_e)
      }
    }
  }
}

// ---------------------------------------------------------------------------
// Transposed conv (ConvTranspose2d k=4,s=2,p=1), parity-decomposed, NHWC in.
// K order = (tap, ci).  Double-buffered async B staging.
// ---------------------------------------------------------------------------
template<int NSUB, bool BIAS, bool RELU_OUT, bool STORE_F16, bool OUT_F32>
__global__ __launch_bounds__(BLK) void convt_wmma(
    const _Float16* __restrict__ inh, const _Float16* __restrict__ Afrag4,
    const float* __restrict__ bias,
    _Float16* __restrict__ outh, float* __restrict__ outf,
    int Ci, int ciSh, int Co, int Hi, int Wi, int Ho, int Wo, int Kpad)
{
  constexpr int POS = NSUB * 16;
  constexpr int TPR = BLK / POS;           // staging threads per position
  constexpr int SPT = POS / 32;            // 16B segments per staging thread
  __shared__ _Float16 ldsB[2 * POS * KT];
  const int tid  = threadIdx.x;
  const int lane = tid & 31;
  const int wv   = tid >> 5;

  const int xt = Wo / (32 * NSUB);         // tiles per parity per output row
  int tile = blockIdx.x;
  const int owt = tile % xt;  tile /= xt;
  const int pw  = tile & 1;   tile >>= 1;
  const int oh  = tile % Ho;
  const int n   = tile / Ho;
  const int ph  = oh & 1;
  const int ihb = (oh + ph) >> 1;
  const int j0  = owt * POS;

  const int mt      = blockIdx.y * 4 + wv;
  const int mTiles  = (Co + 15) / 16;
  const bool mvalid = (mt < mTiles);
  const int co_base = mt * 16;
  const int nCh     = Kpad >> 5;

  const size_t clsStride = (size_t)mTiles * nCh * 512;
  const _Float16* afrag = Afrag4 + (size_t)(ph * 2 + pw) * clsStride
                          + (size_t)mt * nCh * 512 + lane * 16;

  const int nn = lane & 15;
  const int nnS = tid / TPR;
  const int segBase = (tid % TPR) * SPT;
  const int ciMask = (1 << ciSh) - 1;

  v8f c[NSUB];
  #pragma unroll
  for (int s = 0; s < NSUB; ++s) c[s] = (v8f){};

  auto issueChunk = [&](int ck, int bsel) -> bool {
    const int kb  = ck * KT;
    const int tap = kb >> ciSh;
    const int th  = (tap >> 1) & 1;
    const int tw  = tap & 1;
    const int ci0 = kb & ciMask;
    const int ih  = ihb - th;
    const int iw  = j0 + nnS + pw - tw;
    const bool ok = (ih >= 0 && ih < Hi && iw >= 0 && iw < Wi);
    const int ihc = ok ? ih : 0;
    const int iwc = ok ? iw : 0;
    const _Float16* src =
        inh + (((size_t)n * Hi + ihc) * Wi + iwc) * Ci + ci0;
    _Float16* dstRow = &ldsB[bsel * (POS * KT) + nnS * KT];
    #pragma unroll
    for (int u = 0; u < SPT; ++u) {
      const int seg = segBase + u;
      async_b128((uint32_t)(uintptr_t)(dstRow + seg * 8),
                 (const void*)(src + seg * 8));
    }
    return ok;
  };

  bool okCur = issueChunk(0, 0);
  for (int cIdx = 0; cIdx < nCh; ++cIdx) {
    const int cur = cIdx & 1;
    bool okNext = true;
    if (cIdx + 1 < nCh) { okNext = issueChunk(cIdx + 1, cur ^ 1); wait_async<SPT>(); }
    else                { wait_async<0>(); }
    if (!okCur) {
      v8h zz = {};
      #pragma unroll
      for (int u = 0; u < SPT; ++u)
        *(v8h*)(&ldsB[cur * (POS * KT) + nnS * KT + (segBase + u) * 8]) = zz;
    }
    __syncthreads();
    if (mvalid) {
      const v16h a = *(const v16h*)(afrag + (size_t)cIdx * 512);
      const int bo = (lane >= 16) ? 16 : 0;
      #pragma unroll
      for (int s = 0; s < NSUB; ++s) {
        v16h b;
        const _Float16* brow = &ldsB[cur * (POS * KT) + (s * 16 + nn) * KT + bo];
        #pragma unroll
        for (int j = 0; j < 16; ++j) b[j] = brow[j];
        c[s] = __builtin_amdgcn_wmma_f32_16x16x32_f16(false, a, false, b,
                                                      (short)0, c[s], false, false);
      }
    }
    __syncthreads();
    okCur = okNext;
  }

  if (!mvalid) return;
  const int rb = (lane >= 16) ? 8 : 0;
  #pragma unroll
  for (int s = 0; s < NSUB; ++s) {
    const int ow = 2 * (j0 + s * 16 + nn) + pw;
    #pragma unroll
    for (int r = 0; r < 8; ++r) {
      const int com = co_base + rb + r;
      if (com < Co) {
        float v = c[s][r];
        if (BIAS) v += bias[com];
        if (RELU_OUT) v = fmaxf(v, 0.f);
        if (STORE_F16) {
          const size_t oidx = (((size_t)n * Ho + oh) * Wo + ow) * Co + com;
          outh[oidx] = (_Float16)v;                   // NHWC
        }
        if (OUT_F32) {
          const size_t oidx = (((size_t)n * Co + com) * Ho + oh) * Wo + ow;
          outf[oidx] = v;                             // NCHW f32 (final y)
        }
      }
    }
  }
}

// ---------------------------------------------------------------------------
// Packing / conversion kernels
// ---------------------------------------------------------------------------
__global__ void nchw_to_nhwc_f16(const float* __restrict__ in,
                                 _Float16* __restrict__ out,
                                 int C, int HW, size_t tot) {
  size_t i = (size_t)blockIdx.x * blockDim.x + threadIdx.x;
  if (i >= tot) return;
  int c   = (int)(i % C);
  size_t pos = i / C;
  size_t n  = pos / HW;
  size_t hw = pos % HW;
  out[i] = (_Float16)in[(n * C + c) * HW + hw];
}

// OIHW fp32 -> fragment-ordered f16 A: [mt][chunk][lane][16]
// K = (kh*KW + kw)*CiPad + ci
__global__ void pack_conv_frag(const float* __restrict__ w, _Float16* __restrict__ Af,
                               int Co, int CiReal, int CiPad, int KH, int KW,
                               int nCh, size_t tot) {
  size_t i = (size_t)blockIdx.x * blockDim.x + threadIdx.x;
  if (i >= tot) return;
  int j    = (int)(i & 15);
  int lane = (int)((i >> 4) & 31);
  size_t rest = i >> 9;
  int ch = (int)(rest % nCh);
  int mt = (int)(rest / nCh);
  int K  = ch * 32 + kmapA(lane, j);
  int co = mt * 16 + (lane & 15);
  int tap = K / CiPad;
  int ci  = K - tap * CiPad;
  float v = 0.f;
  if (co < Co && ci < CiReal && tap < KH * KW) {
    int kh = tap / KW, kw = tap - kh * KW;
    v = w[(((size_t)co * CiReal + ci) * KH + kh) * KW + kw];
  }
  Af[i] = (_Float16)v;
}

// torch ConvTranspose2d weights (Ci,Co,4,4) -> 4 parity classes, frag order.
__global__ void pack_convt_frag(const float* __restrict__ w, _Float16* __restrict__ Af,
                                int Ci, int Co, int nCh, size_t per) {
  size_t i = (size_t)blockIdx.x * blockDim.x + threadIdx.x;
  if (i >= 4 * per) return;
  int cls = (int)(i / per);
  size_t r = i % per;
  int j    = (int)(r & 15);
  int lane = (int)((r >> 4) & 31);
  size_t rest = r >> 9;
  int ch = (int)(rest % nCh);
  int mt = (int)(rest / nCh);
  int K  = ch * 32 + kmapA(lane, j);
  int co = mt * 16 + (lane & 15);
  int tap = K / Ci;
  int ci  = K - tap * Ci;
  int ph = cls >> 1, pw = cls & 1;
  float v = 0.f;
  if (co < Co && tap < 4) {
    int th = (tap >> 1) & 1, tw = tap & 1;
    int kh = (1 - ph) + 2 * th;
    int kw = (1 - pw) + 2 * tw;
    v = w[(((size_t)ci * Co + co) * 4 + kh) * 4 + kw];
  }
  Af[i] = (_Float16)v;
}

// ---------------------------------------------------------------------------
// Vector quantizer (NHWC): per pixel argmin over 512 codes (dim 64).
// Codebook double-buffered via async LDS DMA (2 x 32 KB ping-pong).
// Deterministic q_loss: per-block LDS reduction + single-block final reduce.
// ---------------------------------------------------------------------------
__global__ __launch_bounds__(256) void vq_kernel(
    const float* __restrict__ ze, const float* __restrict__ emb,
    _Float16* __restrict__ zq, float* __restrict__ partial, int BHW)
{
  __shared__ float semb[2 * 128 * 64];
  __shared__ float red[256];
  const int tid = threadIdx.x;
  const int p   = blockIdx.x * 256 + tid;
  const bool ok = (p < BHW);

  float x[64];
  if (ok) {
    const float* zp = ze + (size_t)p * 64;          // NHWC: contiguous pixel
    #pragma unroll
    for (int c0 = 0; c0 < 64; ++c0) x[c0] = zp[c0];
  }

  auto issuePass = [&](int ps, int bsel) {
    const float* src = emb + (size_t)ps * 8192;
    float* dstBase = &semb[bsel * 8192];
    #pragma unroll
    for (int it = 0; it < 8; ++it) {
      const int idx = it * 256 + tid;               // 16-byte units
      async_b128((uint32_t)(uintptr_t)(dstBase + idx * 4),
                 (const void*)(src + idx * 4));
    }
  };

  float best = 3.4e38f;
  int bi = 0;
  issuePass(0, 0);
  for (int pass = 0; pass < 4; ++pass) {
    const int cur = pass & 1;
    if (pass + 1 < 4) { issuePass(pass + 1, cur ^ 1); wait_async<8>(); }
    else              { wait_async<0>(); }
    __syncthreads();
    if (ok) {
      const float* sb = &semb[cur * 8192];
      for (int k = 0; k < 128; ++k) {
        const float* e = &sb[k * 64];
        float d = 0.f;
        #pragma unroll
        for (int c0 = 0; c0 < 64; ++c0) {
          float t = x[c0] - e[c0];
          d = fmaf(t, t, d);
        }
        if (d < best) { best = d; bi = pass * 128 + k; }
      }
    }
    __syncthreads();
  }

  if (ok) {
    const float* e = emb + (size_t)bi * 64;
    _Float16* zp = zq + (size_t)p * 64;             // NHWC: contiguous pixel
    #pragma unroll
    for (int c0 = 0; c0 < 64; ++c0) zp[c0] = (_Float16)e[c0];
  }

  red[tid] = ok ? best : 0.f;
  __syncthreads();
  for (int st = 128; st > 0; st >>= 1) {
    if (tid < st) red[tid] += red[tid + st];
    __syncthreads();
  }
  if (tid == 0) partial[blockIdx.x] = red[0];
}

__global__ __launch_bounds__(128) void vq_reduce(
    const float* __restrict__ partial, float* __restrict__ qloss,
    int n, float scale)
{
  __shared__ float red[128];
  float s = 0.f;
  for (int i = threadIdx.x; i < n; i += 128) s += partial[i];
  red[threadIdx.x] = s;
  __syncthreads();
  for (int st = 64; st > 0; st >>= 1) {
    if (threadIdx.x < st) red[threadIdx.x] += red[threadIdx.x + st];
    __syncthreads();
  }
  if (threadIdx.x == 0) *qloss = red[0] * scale;
}

// ---------------------------------------------------------------------------
// Host orchestration
// ---------------------------------------------------------------------------
static inline size_t cdivz(size_t a, size_t b) { return (a + b - 1) / b; }

static inline dim3 convGrid(int N, int Ho, int Wo, int Co) {
  int gx = N * Ho * Wo / 64;
  int gy = (((Co + 15) / 16) + 3) / 4;
  return dim3((unsigned)gx, (unsigned)gy, 1);
}
static inline dim3 convtGrid(int N, int Ho, int Wo, int Co, int nsub) {
  int gx = N * Ho * (Wo / (32 * nsub)) * 2;
  int gy = (((Co + 15) / 16) + 3) / 4;
  return dim3((unsigned)gx, (unsigned)gy, 1);
}

extern "C" void kernel_launch(void* const* d_in, const int* in_sizes, int n_in,
                              void* d_out, int out_size, void* d_ws, size_t ws_size,
                              hipStream_t stream) {
  (void)in_sizes; (void)n_in; (void)ws_size;

  const float* x          = (const float*)d_in[0];
  const float* enc_w1     = (const float*)d_in[1];
  const float* enc_b1     = (const float*)d_in[2];
  const float* enc_w2     = (const float*)d_in[3];
  const float* enc_b2     = (const float*)d_in[4];
  const float* enc_w3     = (const float*)d_in[5];
  const float* enc_b3     = (const float*)d_in[6];
  const float* enc_res_w3 = (const float*)d_in[7];
  const float* enc_res_w1 = (const float*)d_in[8];
  const float* prevq_w    = (const float*)d_in[9];
  const float* prevq_b    = (const float*)d_in[10];
  const float* emb        = (const float*)d_in[11];
  const float* dec_w1     = (const float*)d_in[12];
  const float* dec_b1     = (const float*)d_in[13];
  const float* dec_res_w3 = (const float*)d_in[14];
  const float* dec_res_w1 = (const float*)d_in[15];
  const float* dec_w2     = (const float*)d_in[16];
  const float* dec_b2     = (const float*)d_in[17];
  const float* dec_w3     = (const float*)d_in[18];
  const float* dec_b3     = (const float*)d_in[19];

  // --- bump allocator over workspace ---
  char* ws = (char*)d_ws;
  size_t off = 0;
  auto alloc_h = [&](size_t elems) -> _Float16* {
    _Float16* p = (_Float16*)(ws + off);
    off += ((elems * 2 + 255) & ~(size_t)255);
    return p;
  };
  auto alloc_f = [&](size_t elems) -> float* {
    float* p = (float*)(ws + off);
    off += ((elems * 4 + 255) & ~(size_t)255);
    return p;
  };

  // packed weights (f16, fragment order). size = Copad*Kpad (x4 for convT)
  _Float16* wpE1  = alloc_h((size_t)128 * 64);        // CiPad=4,  Kpad=64
  _Float16* wpE2  = alloc_h((size_t)256 * 2048);
  _Float16* wpE3  = alloc_h((size_t)512 * 4096);
  _Float16* wpER3 = alloc_h((size_t)32 * 4608);
  _Float16* wpER1 = alloc_h((size_t)512 * 32);
  _Float16* wpPQ  = alloc_h((size_t)64 * 512);
  _Float16* wpD1  = alloc_h((size_t)4 * 256 * 256);
  _Float16* wpDR3 = alloc_h((size_t)32 * 2304);
  _Float16* wpDR1 = alloc_h((size_t)256 * 32);
  _Float16* wpD2  = alloc_h((size_t)4 * 128 * 1024);
  _Float16* wpD3  = alloc_h((size_t)4 * 16 * 512);

  // activations (NHWC f16), ping-ponged
  _Float16* xh   = alloc_h((size_t)32 * 256 * 256 * 3);   // x NHWC
  _Float16* bufA = alloc_h((size_t)32 * 128 * 128 * 128); // z1 / y2
  _Float16* bufB = alloc_h((size_t)32 * 64 * 64 * 256);   // z2 / y1
  _Float16* bufC = alloc_h((size_t)32 * 32 * 32 * 512);   // z3
  _Float16* tmpR = alloc_h((size_t)32 * 64 * 64 * 32);    // res hidden (max)
  float*    zef  = alloc_f((size_t)32 * 32 * 32 * 64);    // z_e NHWC fp32
  _Float16* zqh  = alloc_h((size_t)32 * 32 * 32 * 64);    // z_q NHWC f16
  float*    part = alloc_f(128);

  // --- input conversion + weight packing ---
  {
    size_t n = (size_t)32 * 256 * 256 * 3;
    nchw_to_nhwc_f16<<<(unsigned)cdivz(n, 256), 256, 0, stream>>>(x, xh, 3, 256 * 256, n);
  }
  pack_conv_frag<<<(unsigned)cdivz((size_t)128 * 64, 256), 256, 0, stream>>>(
      enc_w1, wpE1, 128, 3, 4, 4, 4, 64 / 32, (size_t)128 * 64);
  pack_conv_frag<<<(unsigned)cdivz((size_t)256 * 2048, 256), 256, 0, stream>>>(
      enc_w2, wpE2, 256, 128, 128, 4, 4, 2048 / 32, (size_t)256 * 2048);
  pack_conv_frag<<<(unsigned)cdivz((size_t)512 * 4096, 256), 256, 0, stream>>>(
      enc_w3, wpE3, 512, 256, 256, 4, 4, 4096 / 32, (size_t)512 * 4096);
  pack_conv_frag<<<(unsigned)cdivz((size_t)32 * 4608, 256), 256, 0, stream>>>(
      enc_res_w3, wpER3, 32, 512, 512, 3, 3, 4608 / 32, (size_t)32 * 4608);
  pack_conv_frag<<<(unsigned)cdivz((size_t)512 * 32, 256), 256, 0, stream>>>(
      enc_res_w1, wpER1, 512, 32, 32, 1, 1, 1, (size_t)512 * 32);
  pack_conv_frag<<<(unsigned)cdivz((size_t)64 * 512, 256), 256, 0, stream>>>(
      prevq_w, wpPQ, 64, 512, 512, 1, 1, 512 / 32, (size_t)64 * 512);
  pack_convt_frag<<<(unsigned)cdivz((size_t)4 * 256 * 256, 256), 256, 0, stream>>>(
      dec_w1, wpD1, 64, 256, 256 / 32, (size_t)256 * 256);
  pack_conv_frag<<<(unsigned)cdivz((size_t)32 * 2304, 256), 256, 0, stream>>>(
      dec_res_w3, wpDR3, 32, 256, 256, 3, 3, 2304 / 32, (size_t)32 * 2304);
  pack_conv_frag<<<(unsigned)cdivz((size_t)256 * 32, 256), 256, 0, stream>>>(
      dec_res_w1, wpDR1, 256, 32, 32, 1, 1, 1, (size_t)256 * 32);
  pack_convt_frag<<<(unsigned)cdivz((size_t)4 * 128 * 1024, 256), 256, 0, stream>>>(
      dec_w2, wpD2, 256, 128, 1024 / 32, (size_t)128 * 1024);
  pack_convt_frag<<<(unsigned)cdivz((size_t)4 * 16 * 512, 256), 256, 0, stream>>>(
      dec_w3, wpD3, 128, 3, 512 / 32, (size_t)16 * 512);

  // --- encoder ---
  conv_wmma<4,4,2,1, false,false, true,true,false,true,false>
    <<<convGrid(32,128,128,128), BLK, 0, stream>>>(
      xh, wpE1, enc_b1, nullptr, bufA, nullptr,
      3, 2, 128, 256, 256, 128, 128, 64, 14, 7);
  conv_wmma<4,4,2,1, true,false, true,true,false,true,false>
    <<<convGrid(32,64,64,256), BLK, 0, stream>>>(
      bufA, wpE2, enc_b2, nullptr, bufB, nullptr,
      128, 7, 256, 128, 128, 64, 64, 2048, 12, 6);
  conv_wmma<4,4,2,1, true,false, true,false,false,true,false>
    <<<convGrid(32,32,32,512), BLK, 0, stream>>>(
      bufB, wpE3, enc_b3, nullptr, bufC, nullptr,
      256, 8, 512, 64, 64, 32, 32, 4096, 10, 5);

  // encoder res stack (shared weights, 2 layers).
  // conv3x3 stores relu(h) so the 1x1 can use the async path.
  conv_wmma<3,3,1,1, false,true, false,true,false,true,false>
    <<<convGrid(32,32,32,32), BLK, 0, stream>>>(
      bufC, wpER3, nullptr, nullptr, tmpR, nullptr,
      512, 9, 32, 32, 32, 32, 32, 4608, 10, 5);
  conv_wmma<1,1,1,0, true,false, false,false,true,true,false>
    <<<convGrid(32,32,32,512), BLK, 0, stream>>>(
      tmpR, wpER1, nullptr, bufC, bufC, nullptr,
      32, 5, 512, 32, 32, 32, 32, 32, 10, 5);
  conv_wmma<3,3,1,1, false,true, false,true,false,true,false>
    <<<convGrid(32,32,32,32), BLK, 0, stream>>>(
      bufC, wpER3, nullptr, nullptr, tmpR, nullptr,
      512, 9, 32, 32, 32, 32, 32, 4608, 10, 5);
  conv_wmma<1,1,1,0, true,false, false,true,true,true,false>
    <<<convGrid(32,32,32,512), BLK, 0, stream>>>(
      tmpR, wpER1, nullptr, bufC, bufC, nullptr,
      32, 5, 512, 32, 32, 32, 32, 32, 10, 5);

  // pre-VQ 1x1 conv -> z_e NHWC fp32
  conv_wmma<1,1,1,0, true,false, true,false,false,false,true>
    <<<convGrid(32,32,32,64), BLK, 0, stream>>>(
      bufC, wpPQ, prevq_b, nullptr, nullptr, zef,
      512, 9, 64, 32, 32, 32, 32, 512, 10, 5);

  // --- vector quantizer + q_loss ---
  vq_kernel<<<128, 256, 0, stream>>>(zef, emb, zqh, part, 32768);
  vq_reduce<<<1, 128, 0, stream>>>(part, (float*)d_out + (out_size - 1), 128,
                                   2.0f / (32.0f * 1024.0f * 64.0f));

  // --- decoder ---
  convt_wmma<2, true,false,true,false>
    <<<convtGrid(32,64,64,256,2), BLK, 0, stream>>>(
      zqh, wpD1, dec_b1, bufB, nullptr, 64, 6, 256, 32, 32, 64, 64, 256);

  // decoder res stack
  conv_wmma<3,3,1,1, false,true, false,true,false,true,false>
    <<<convGrid(32,64,64,32), BLK, 0, stream>>>(
      bufB, wpDR3, nullptr, nullptr, tmpR, nullptr,
      256, 8, 32, 64, 64, 64, 64, 2304, 12, 6);
  conv_wmma<1,1,1,0, true,false, false,false,true,true,false>
    <<<convGrid(32,64,64,256), BLK, 0, stream>>>(
      tmpR, wpDR1, nullptr, bufB, bufB, nullptr,
      32, 5, 256, 64, 64, 64, 64, 32, 12, 6);
  conv_wmma<3,3,1,1, false,true, false,true,false,true,false>
    <<<convGrid(32,64,64,32), BLK, 0, stream>>>(
      bufB, wpDR3, nullptr, nullptr, tmpR, nullptr,
      256, 8, 32, 64, 64, 64, 64, 2304, 12, 6);
  conv_wmma<1,1,1,0, true,false, false,true,true,true,false>
    <<<convGrid(32,64,64,256), BLK, 0, stream>>>(
      tmpR, wpDR1, nullptr, bufB, bufB, nullptr,
      32, 5, 256, 64, 64, 64, 64, 32, 12, 6);

  convt_wmma<4, true,true,true,false>
    <<<convtGrid(32,128,128,128,4), BLK, 0, stream>>>(
      bufB, wpD2, dec_b2, bufA, nullptr, 256, 8, 128, 64, 64, 128, 128, 1024);

  convt_wmma<4, true,false,false,true>
    <<<convtGrid(32,256,256,3,4), BLK, 0, stream>>>(
      bufA, wpD3, dec_b3, nullptr, (float*)d_out, 128, 7, 3, 128, 128, 256, 256, 512);
}
